// Fusion_25924422598803
// MI455X (gfx1250) — matrix operand order repaired and checked by reading, not compiled, not run on hardware
//
#include <hip/hip_runtime.h>
#include <hip/hip_bf16.h>

#define DEVI __device__ __forceinline__

typedef __bf16 bf16;
typedef __attribute__((ext_vector_type(16))) __bf16 bf16x16;
typedef __attribute__((ext_vector_type(8)))  __bf16 bf16x8;
typedef __attribute__((ext_vector_type(8)))  float  f32x8;
typedef __attribute__((ext_vector_type(4)))  float  f32x4;

constexpr int   Bsz   = 8;
constexpr int   C     = 256;
constexpr int   S     = 2304;     // 48*48
constexpr float EPS   = 1e-5f;
constexpr float SCALE = 0.0625f;  // 1/sqrt(256)

// ---------------- WMMA fragment helpers (wave32, 16x16x32 bf16) ----------------
// A fragment (M=16 x K=32), source row-major [m][k] with leading dim ld.
// ISA layout: lanes 0-15 -> M=0..15, K in {0..7,16..23}; lanes 16-31 -> K in {8..15,24..31}.
DEVI bf16x16 load_frag_a(const bf16* base, int row0, int k0, int ld) {
  const int lane = threadIdx.x & 31;
  const int m    = lane & 15;
  const int kh   = (lane >> 4) << 3;          // 0 or 8
  const bf16* p  = base + (size_t)(row0 + m) * ld + k0 + kh;
  bf16x8 lo = *(const bf16x8*)(p);            // K = k0+kh+0..7
  bf16x8 hi = *(const bf16x8*)(p + 16);       // K = k0+16+kh+0..7
  bf16x16 f;
#pragma unroll
  for (int i = 0; i < 8; ++i) { f[i] = lo[i]; f[i + 8] = hi[i]; }
  return f;
}

// B fragment (K=32 x N=16) from a transposed ("BT") matrix stored row-major as [n][k].
// ISA layout: lane n holds its column; lanes 0-15 K in {0..7,16..23}, lanes 16-31 K in {8..15,24..31}.
DEVI bf16x16 load_frag_bt(const bf16* bt, int n0, int k0, int ld) {
  const int lane = threadIdx.x & 31;
  const int n    = lane & 15;
  const int kh   = (lane >> 4) << 3;          // 0 or 8
  const bf16* p  = bt + (size_t)(n0 + n) * ld + k0 + kh;
  bf16x8 lo = *(const bf16x8*)(p);
  bf16x8 hi = *(const bf16x8*)(p + 16);
  bf16x16 f;
#pragma unroll
  for (int i = 0; i < 8; ++i) { f[i] = lo[i]; f[i + 8] = hi[i]; }
  return f;
}

DEVI f32x8 wmma_bf16(bf16x16 a, bf16x16 b, f32x8 c) {
  return __builtin_amdgcn_wmma_f32_16x16x32_bf16(false, a, false, b, (short)0, c, false, false);
}

// ---------------- Kernel 1: f32 weights -> bf16 transposed [n][k] ----------------
__global__ void wt_transpose_kernel(const float* __restrict__ Wq, const float* __restrict__ Wk,
                                    const float* __restrict__ Wv, const float* __restrict__ Wo,
                                    bf16* __restrict__ WT) {
  const float* W = (blockIdx.x == 0) ? Wq : (blockIdx.x == 1) ? Wk
                  : (blockIdx.x == 2) ? Wv : Wo;
  bf16* T = WT + (size_t)blockIdx.x * C * C;
  __shared__ float tile[32][33];
  const int ty = threadIdx.x >> 5, tx = threadIdx.x & 31;
  for (int tid = 0; tid < 64; ++tid) {
    const int tr = (tid >> 3) * 32;   // k block
    const int tc = (tid & 7) * 32;    // n block
#pragma unroll
    for (int i = 0; i < 4; ++i)
      tile[ty + i * 8][tx] = W[(size_t)(tr + ty + i * 8) * C + tc + tx];
    __syncthreads();
#pragma unroll
    for (int i = 0; i < 4; ++i)
      T[(size_t)(tc + ty + i * 8) * C + tr + tx] = (bf16)tile[tx][ty + i * 8];
    __syncthreads();
  }
}

// ---------------- Kernel 2: LayerNorm over C, coalesced via LDS tile ----------------
// x is [B][C][S]; output xln is bf16 [2][B*S][C] row-major.
__global__ void layernorm_kernel(const float* __restrict__ x_l, const float* __restrict__ x_r,
                                 const float* __restrict__ g1, const float* __restrict__ b1,
                                 const float* __restrict__ g2, const float* __restrict__ b2,
                                 bf16* __restrict__ xln) {
  const int side = blockIdx.y;
  const int b    = blockIdx.x / (S / 32);
  const int s0   = (blockIdx.x % (S / 32)) * 32;
  const float* x  = side ? x_r : x_l;
  const float* g  = side ? g2 : g1;
  const float* be = side ? b2 : b1;
  bf16* out = xln + ((size_t)side * Bsz * S + (size_t)b * S) * C;

  __shared__ float tile[C][33];
  __shared__ float psum[8][32], psq[8][32];
  __shared__ float mu[32], rs[32];

  const int t = threadIdx.x;
  {
    const float* src = x + ((size_t)b * C + t) * S + s0;
#pragma unroll 4
    for (int i = 0; i < 32; ++i) tile[t][i] = src[i];
  }
  __syncthreads();
  {
    const int sl = t & 31, part = t >> 5;
    float s1 = 0.f, s2 = 0.f;
#pragma unroll
    for (int j = 0; j < 32; ++j) {
      float v = tile[part * 32 + j][sl];
      s1 += v; s2 += v * v;
    }
    psum[part][sl] = s1; psq[part][sl] = s2;
  }
  __syncthreads();
  if (t < 32) {
    float s1 = 0.f, s2 = 0.f;
#pragma unroll
    for (int p = 0; p < 8; ++p) { s1 += psum[p][t]; s2 += psq[p][t]; }
    const float m = s1 * (1.0f / C);
    const float v = s2 * (1.0f / C) - m * m;
    mu[t] = m;
    rs[t] = rsqrtf(v + EPS);
  }
  __syncthreads();
  {
    const int s  = t >> 3;
    const int c0 = (t & 7) * 32;
    const float m = mu[s], r = rs[s];
    bf16* dst = out + (size_t)(s0 + s) * C + c0;
#pragma unroll 4
    for (int j = 0; j < 32; ++j) {
      const int c = c0 + j;
      dst[j] = (bf16)((tile[c][s] - m) * r * g[c] + be[c]);
    }
  }
}

// ---------------- Kernel 3: Q/K/V projections via WMMA ----------------
// Q,K: bf16 [2][B*S][C] row-major.  V written transposed: [2][B][C][S].
__global__ void qkv_gemm_kernel(const bf16* __restrict__ xln, const bf16* __restrict__ WT,
                                bf16* __restrict__ Qb, bf16* __restrict__ Kb,
                                bf16* __restrict__ Vt) {
  const int row0  = blockIdx.x * 16;
  const int which = blockIdx.y % 3;       // 0=Q 1=K 2=V
  const int side  = blockIdx.y / 3;
  const int wave  = threadIdx.x >> 5;     // 4 waves: 64 output cols each
  const int lane  = threadIdx.x & 31;
  const int c0    = wave * 64;

  const bf16* A  = xln + (size_t)side * Bsz * S * C;
  const bf16* BT = WT + (size_t)which * C * C;

  f32x8 acc[4] = {};
#pragma unroll
  for (int kk = 0; kk < 8; ++kk) {
    bf16x16 a = load_frag_a(A, row0, kk * 32, C);
#pragma unroll
    for (int j = 0; j < 4; ++j) {
      bf16x16 bt = load_frag_bt(BT, c0 + j * 16, kk * 32, C);
      acc[j] = wmma_bf16(a, bt, acc[j]);
    }
  }

  const int hi8 = (lane >> 4) << 3;
  if (which < 2) {
    bf16* out = (which == 0 ? Qb : Kb) + (size_t)side * Bsz * S * C;
#pragma unroll
    for (int j = 0; j < 4; ++j) {
      const int c = c0 + j * 16 + (lane & 15);
#pragma unroll
      for (int r = 0; r < 8; ++r)
        out[(size_t)(row0 + hi8 + r) * C + c] = (bf16)acc[j][r];
    }
  } else {
    const int b  = row0 / S;
    const int sb = row0 % S;
    bf16* out = Vt + ((size_t)side * Bsz + b) * C * S;
#pragma unroll
    for (int j = 0; j < 4; ++j) {
      const int c = c0 + j * 16 + (lane & 15);
      bf16x8 v;
#pragma unroll
      for (int r = 0; r < 8; ++r) v[r] = (bf16)acc[j][r];
      *(bf16x8*)(out + (size_t)c * S + sb + hi8) = v;   // contiguous along s
    }
  }
}

// ---------------- Kernel 4: fused cross-batch-softmax attention ----------------
// Block = 8 waves; wave b handles batch b for a 16-row s-tile. Softmax is over
// the batch axis (dim 0), so it is point-wise across the 8 waves via LDS.
__global__ void __launch_bounds__(256)
attention_kernel(const bf16* __restrict__ Qb, const bf16* __restrict__ Kb,
                 const bf16* __restrict__ Vt, bf16* __restrict__ attn) {
  const int side = blockIdx.y;
  const int s0   = blockIdx.x * 16;
  const int b    = threadIdx.x >> 5;      // wave id == batch index
  const int lane = threadIdx.x & 31;

  const bf16* Q = Qb + ((size_t)side * Bsz + b) * S * C;        // this side
  const bf16* K = Kb + ((size_t)(1 - side) * Bsz + b) * S * C;  // cross side
  const bf16* V = Vt + ((size_t)side * Bsz + b) * C * S;        // this side, transposed
  bf16* Ao = attn + ((size_t)side * Bsz + b) * S * C;

  __shared__ float sS[Bsz][16][32];
  __shared__ bf16  sP[Bsz][16][32];

  // Q panel (16 x 256) resident in registers: 8 A-fragments
  bf16x16 qf[8];
#pragma unroll
  for (int kk = 0; kk < 8; ++kk) qf[kk] = load_frag_a(Q, s0, kk * 32, C);

  f32x8 acc[16] = {};                     // out rows 16 x 256 fp32
  const int hi8 = (lane >> 4) << 3;

  for (int t0 = 0; t0 < S; t0 += 32) {
    // scores: 16x32 = Qpanel @ K[t0:t0+32,:]^T   (16 WMMA)
    f32x8 sc0 = {}, sc1 = {};
#pragma unroll
    for (int kk = 0; kk < 8; ++kk) {
      bf16x16 bk0 = load_frag_bt(K, t0,      kk * 32, C);
      bf16x16 bk1 = load_frag_bt(K, t0 + 16, kk * 32, C);
      sc0 = wmma_bf16(qf[kk], bk0, sc0);
      sc1 = wmma_bf16(qf[kk], bk1, sc1);
    }
#pragma unroll
    for (int r = 0; r < 8; ++r) {
      sS[b][hi8 + r][lane & 15]        = sc0[r] * SCALE;
      sS[b][hi8 + r][16 + (lane & 15)] = sc1[r] * SCALE;
    }
    __syncthreads();
    // softmax over the 8 batches for each of the 512 (s,t) positions
#pragma unroll
    for (int p = 0; p < 2; ++p) {
      const int pos = threadIdx.x + p * 256;
      const int sr = pos >> 5, tc = pos & 31;
      float v[8], mx = -3.4e38f;
#pragma unroll
      for (int bb = 0; bb < 8; ++bb) { v[bb] = sS[bb][sr][tc]; mx = fmaxf(mx, v[bb]); }
      float sum = 0.f;
#pragma unroll
      for (int bb = 0; bb < 8; ++bb) { v[bb] = __expf(v[bb] - mx); sum += v[bb]; }
      const float inv = 1.0f / sum;
#pragma unroll
      for (int bb = 0; bb < 8; ++bb) sP[bb][sr][tc] = (bf16)(v[bb] * inv);
    }
    __syncthreads();
    // acc += P(16x32) @ V[t0:t0+32, :]   (16 WMMA, B-frags contiguous from V^T)
    bf16x16 pf = load_frag_a(&sP[b][0][0], 0, 0, 32);
#pragma unroll
    for (int n = 0; n < 16; ++n) {
      bf16x16 bv = load_frag_bt(V, n * 16, t0, S);
      acc[n] = wmma_bf16(pf, bv, acc[n]);
    }
    __syncthreads();
  }
  // store attention output (pre-Wo) bf16 row-major [s][c]
#pragma unroll
  for (int n = 0; n < 16; ++n) {
    const int c = n * 16 + (lane & 15);
#pragma unroll
    for (int r = 0; r < 8; ++r)
      Ao[(size_t)(s0 + hi8 + r) * C + c] = (bf16)acc[n][r];
  }
}

// ---------------- Kernel 5: out = x + attn @ Wo, written as [B][C][S] ----------------
__global__ void out_gemm_kernel(const bf16* __restrict__ attn, const bf16* __restrict__ WoT,
                                const float* __restrict__ x_l, const float* __restrict__ x_r,
                                float* __restrict__ out) {
  const int side = blockIdx.y;
  const int row0 = blockIdx.x * 16;
  const int wave = threadIdx.x >> 5;
  const int lane = threadIdx.x & 31;
  const int c0   = wave * 64;

  const bf16* A  = attn + (size_t)side * Bsz * S * C;
  const float* x = side ? x_r : x_l;
  float* o = out + (size_t)side * Bsz * C * S;

  f32x8 acc[4] = {};
#pragma unroll
  for (int kk = 0; kk < 8; ++kk) {
    bf16x16 a = load_frag_a(A, row0, kk * 32, C);
#pragma unroll
    for (int j = 0; j < 4; ++j) {
      bf16x16 bt = load_frag_bt(WoT, c0 + j * 16, kk * 32, C);
      acc[j] = wmma_bf16(a, bt, acc[j]);
    }
  }
  const int b  = row0 / S;
  const int sb = (row0 % S) + ((lane >> 4) << 3);
#pragma unroll
  for (int j = 0; j < 4; ++j) {
    const int c = c0 + j * 16 + (lane & 15);
    const size_t off = ((size_t)b * C + c) * S + sb;   // contiguous along s
    f32x4 xr0 = *(const f32x4*)(x + off);
    f32x4 xr1 = *(const f32x4*)(x + off + 4);
    f32x4 o0, o1;
#pragma unroll
    for (int r = 0; r < 4; ++r) { o0[r] = xr0[r] + acc[j][r]; o1[r] = xr1[r] + acc[j][r + 4]; }
    *(f32x4*)(o + off)     = o0;
    *(f32x4*)(o + off + 4) = o1;
  }
}

// ---------------- host launcher ----------------
extern "C" void kernel_launch(void* const* d_in, const int* in_sizes, int n_in,
                              void* d_out, int out_size, void* d_ws, size_t ws_size,
                              hipStream_t stream) {
  (void)in_sizes; (void)n_in; (void)out_size; (void)ws_size;
  const float* x_l = (const float*)d_in[0];
  const float* x_r = (const float*)d_in[1];
  const float* Wq  = (const float*)d_in[2];
  const float* Wk  = (const float*)d_in[3];
  const float* Wv  = (const float*)d_in[4];
  const float* Wo  = (const float*)d_in[5];
  const float* g1  = (const float*)d_in[6];
  const float* b1  = (const float*)d_in[7];
  const float* g2  = (const float*)d_in[8];
  const float* b2  = (const float*)d_in[9];

  const size_t MAT = (size_t)Bsz * S * C;   // 4,718,592 elems
  bf16* ws   = (bf16*)d_ws;
  bf16* xln  = ws;                // [2][B*S][C]
  bf16* Qb   = ws + 2 * MAT;      // [2][B*S][C]
  bf16* Kb   = ws + 4 * MAT;      // [2][B*S][C]
  bf16* Vt   = ws + 6 * MAT;      // [2][B][C][S]
  bf16* attn = ws + 8 * MAT;      // [2][B*S][C]
  bf16* WT   = ws + 10 * MAT;     // [4][C][C] bf16 transposed weights

  wt_transpose_kernel<<<dim3(4), 256, 0, stream>>>(Wq, Wk, Wv, Wo, WT);
  layernorm_kernel<<<dim3(Bsz * (S / 32), 2), 256, 0, stream>>>(x_l, x_r, g1, b1, g2, b2, xln);
  qkv_gemm_kernel<<<dim3(Bsz * S / 16, 6), 128, 0, stream>>>(xln, WT, Qb, Kb, Vt);
  attention_kernel<<<dim3(S / 16, 2), 256, 0, stream>>>(Qb, Kb, Vt, attn);
  out_gemm_kernel<<<dim3(Bsz * S / 16, 2), 128, 0, stream>>>(
      attn, WT + (size_t)3 * C * C, x_l, x_r, (float*)d_out);
}
